// GraphAttentionLayer_19928648254256
// MI455X (gfx1250) — matrix operand order repaired and checked by reading, not compiled, not run on hardware
//
#include <hip/hip_runtime.h>
#include <hip/hip_bf16.h>

// GAT layer for MI455X (gfx1250, wave32, WMMA + async global->LDS).
// B=8, N=2048, IN_DIM=128, OUT_DIM=64, HEADS=4.
//
// Attention-apply is 17.2 GFLOP vs ~50MB traffic -> compute bound. Softmax is
// streamed; leaky_relu is monotonic so row max = lrelu(Wh1[n] + max_m Wh2[m])
// is known up front. Wh1/Wh2 live in the log2(e)-scaled domain (lrelu commutes
// with positive scale) so p = exp2(...) with no multiply; the -rmax shift and
// the 0.2 lrelu slope are folded into per-row scalars c1/c2 plus a
// precomputed 0.2*Wh2 stream, leaving add/add/max/exp2 per element. Row sums
// ride a 5th WMMA against an all-ones B tile (its C rows r/r+8 are exactly the
// rows each lane normalizes), so no cross-lane reduction is needed.

#define GAT_B 8
#define GAT_N 2048
#define GAT_IN 128
#define GAT_OUT 64
#define GAT_H 4
#define GAT_COLS (GAT_OUT * GAT_H) // 256
#define STAGE_M 64                  // m-values staged per double-buffer slot
#define LOG2E 1.4426950408889634f

typedef __attribute__((ext_vector_type(16))) __bf16 v16bf;
typedef __attribute__((ext_vector_type(2)))  __bf16 v2bf;
typedef __attribute__((ext_vector_type(8)))  float  v8f;
typedef __attribute__((ext_vector_type(4)))  int    v4i_t;
typedef __attribute__((address_space(1)))    v4i_t  as1_v4i;
typedef __attribute__((address_space(3)))    v4i_t  as3_v4i;

__device__ __forceinline__ float lrelu02(float x) {
    return fmaxf(x, 0.2f * x);
}

__device__ __forceinline__ unsigned pack2_bf16(float lo, float hi) {
#if __has_builtin(__builtin_amdgcn_cvt_pk_bf16_f32)
    union { v2bf v; unsigned u; } cv;
    cv.v = __builtin_amdgcn_cvt_pk_bf16_f32(lo, hi);
    return cv.u;
#else
    // gfx1250 VOP3 (asm-verified round 5): one instruction per pair
    unsigned r;
    asm("v_cvt_pk_bf16_f32 %0, %1, %2" : "=v"(r) : "v"(lo), "v"(hi));
    return r;
#endif
}

// ---- CDNA5 async global->LDS (ASYNCcnt) ----------------------------------
template <int N>
__device__ __forceinline__ void wait_async_le() {
#if __has_builtin(__builtin_amdgcn_s_wait_asynccnt)
    __builtin_amdgcn_s_wait_asynccnt(N);
#else
    asm volatile("s_wait_asynccnt %0" :: "n"(N) : "memory");
#endif
}

__device__ __forceinline__ void async_copy_b128(const unsigned short* gsrc,
                                                unsigned short* lds_dst) {
#if __has_builtin(__builtin_amdgcn_global_load_async_to_lds_b128)
    __builtin_amdgcn_global_load_async_to_lds_b128(
        (as1_v4i*)(unsigned long long)(const void*)gsrc,
        (as3_v4i*)(unsigned)(unsigned long long)(void*)lds_dst,
        0, 0);
#else
    asm volatile("global_load_async_to_lds_b128 %0, %1, off"
                 :: "v"((unsigned)(unsigned long long)(void*)lds_dst), "v"(gsrc)
                 : "memory");
#endif
}

// ---------------------------------------------------------------------------
// Kernel 1: Wh = h @ W (per (b,n) row), stored transposed as bf16
// WhT[bh][d][m], plus per-head logits Wh1/Wh2 pre-scaled by log2(e).
// ---------------------------------------------------------------------------
__global__ __launch_bounds__(256) void gat_proj_kernel(
    const float* __restrict__ hsrc, const float* __restrict__ W,
    const float* __restrict__ a,
    unsigned short* __restrict__ WhT,
    float* __restrict__ Wh1, float* __restrict__ Wh2)
{
    __shared__ float hrow[GAT_IN];
    __shared__ float s1[256];
    __shared__ float s2[256];

    const int t  = threadIdx.x;
    const int bn = blockIdx.x;              // b*N + n
    const int b  = bn >> 11;
    const int n  = bn & (GAT_N - 1);

    if (t < GAT_IN) hrow[t] = hsrc[(size_t)bn * GAT_IN + t];
    __syncthreads();

    float acc = 0.f;
#pragma unroll 4
    for (int k = 0; k < GAT_IN; ++k)
        acc = fmaf(hrow[k], W[k * GAT_COLS + t], acc);

    const int hh = t >> 6;                  // head
    const int d  = t & 63;                  // dim in head
    const int bh = b * GAT_H + hh;

    WhT[((size_t)bh * GAT_OUT + d) * GAT_N + n] =
        (unsigned short)(pack2_bf16(acc, 0.f) & 0xFFFFu);

    // per-head dots with a1 (rows 0..63) / a2 (rows 64..127); a is (128,4) row-major
    s1[t] = acc * a[d * GAT_H + hh];
    s2[t] = acc * a[(GAT_OUT + d) * GAT_H + hh];
    __syncthreads();
#pragma unroll
    for (int off = 32; off > 0; off >>= 1) {
        if (d < off) { s1[t] += s1[t + off]; s2[t] += s2[t + off]; }
        __syncthreads();
    }
    if (d == 0) {
        Wh1[(size_t)bh * GAT_N + n] = s1[t] * LOG2E;   // log2-domain logits
        Wh2[(size_t)bh * GAT_N + n] = s2[t] * LOG2E;
    }
}

// ---------------------------------------------------------------------------
// Kernel 2: M2[bh] = max_m Wh2[bh][m] (scaled domain; max commutes), and
// Wh2s = 0.2 * Wh2 (pre-slope stream for the lrelu max-form).
// ---------------------------------------------------------------------------
__global__ __launch_bounds__(256) void gat_max_kernel(
    const float* __restrict__ Wh2, float* __restrict__ M2,
    float* __restrict__ Wh2s)
{
    __shared__ float sm[256];
    const int bh = blockIdx.x;
    float m = -3.4e38f;
    for (int i = threadIdx.x; i < GAT_N; i += 256) {
        const float v = Wh2[(size_t)bh * GAT_N + i];
        Wh2s[(size_t)bh * GAT_N + i] = 0.2f * v;
        m = fmaxf(m, v);
    }
    sm[threadIdx.x] = m;
    __syncthreads();
#pragma unroll
    for (int off = 128; off > 0; off >>= 1) {
        if (threadIdx.x < off)
            sm[threadIdx.x] = fmaxf(sm[threadIdx.x], sm[threadIdx.x + off]);
        __syncthreads();
    }
    if (threadIdx.x == 0) M2[bh] = sm[0];
}

// ---------------------------------------------------------------------------
// Kernel 3: streaming softmax + WMMA apply + ELU.
// 8 waves/block share one (b,h); each wave owns a 16-row n-tile. Per stage the
// block async-copies WhT[bh][0:64][m0:m0+64] (8KB) into one LDS buffer while
// consuming the other. Per K=32 chunk: 4 B tiles preloaded from LDS, 16 exp2s
// (A layout: lanes 0-15 K={0..7,16..23}, lanes 16-31 K={8..15,24..31}), one
// v_cvt_pk_bf16_f32 per pair, then 5 WMMAs: 4 for p@Wh (d=64) + 1 vs all-ones
// B carrying the softmax row sums.
// ---------------------------------------------------------------------------
__global__ __launch_bounds__(256) void gat_attn_kernel(
    const unsigned short* __restrict__ WhT,
    const float* __restrict__ Wh1, const float* __restrict__ Wh2,
    const float* __restrict__ Wh2s,
    const float* __restrict__ M2, float* __restrict__ out)
{
    __shared__ unsigned short stage[2][GAT_OUT * STAGE_M]; // 2 x 8KB

    const int t    = threadIdx.x;
    const int lane = t & 31;
    const int wave = t >> 5;

    const int bh   = blockIdx.x >> 4;                  // 0..31
    const int tile = ((blockIdx.x & 15) << 3) + wave;  // 0..127
    const int n0   = tile << 4;
    const int b    = bh >> 2;
    const int hh   = bh & 3;

    const int row   = lane & 15;       // A row / C column index
    const int hi    = lane >> 4;       // lane-half select
    const int khalf = hi << 3;         // A K sub-offset per layout

    const float* wh2_bh  = Wh2  + (size_t)bh * GAT_N;
    const float* wh2s_bh = Wh2s + (size_t)bh * GAT_N;
    const unsigned short* whT_bh = WhT + (size_t)bh * GAT_OUT * GAT_N;

    // async staging: thread t owns two 16B segments of the 64x64 bf16 slab
    const int srow = t >> 2;            // d row 0..63
    const int sseg = (t & 3) * 2;       // segment pair (8 bf16 each)
    const unsigned short* gseg = whT_bh + (size_t)srow * GAT_N + sseg * 8;
    unsigned short* lseg0 = &stage[0][srow * STAGE_M + sseg * 8];
    unsigned short* lseg1 = &stage[1][srow * STAGE_M + sseg * 8];

    const float wh1n = Wh1[(size_t)bh * GAT_N + n0 + row];
    const float rmax = lrelu02(wh1n + M2[bh]);     // exact row max (monotonicity)
    const float c1 = wh1n - rmax;                  // lrelu(x)-rmax = max(c1+w2, c2+0.2*w2)
    const float c2 = 0.2f * wh1n - rmax;

    // all-ones bf16 B tile for the row-sum WMMA
    union BU { v16bf v; unsigned u[8]; uint4 q[2]; };
    BU Bones;
#pragma unroll
    for (int j = 0; j < 8; ++j) Bones.u[j] = 0x3F803F80u;  // bf16(1.0) x2

    v8f acc[4];
#pragma unroll
    for (int dt = 0; dt < 4; ++dt) acc[dt] = (v8f){};
    v8f accS = (v8f){};                                    // row sums

    // prologue: stage 0
    async_copy_b128(gseg, lseg0);
    async_copy_b128(gseg + 8, lseg0 + 8);

    const int NSTAGE = GAT_N / STAGE_M;            // 32
    for (int s = 0; s < NSTAGE; ++s) {
        const int m0 = s * STAGE_M;
        unsigned short* lnext = ((s + 1) & 1) ? lseg1 : lseg0;
        if (s + 1 < NSTAGE) {
            const unsigned short* gnext = gseg + (m0 + STAGE_M);
            async_copy_b128(gnext, lnext);
            async_copy_b128(gnext + 8, lnext + 8);
            wait_async_le<2>();                    // current stage's copies done
        } else {
            wait_async_le<0>();
        }
        __syncthreads();                           // all waves' copies visible

        const unsigned short* buf = stage[s & 1];
#pragma unroll
        for (int k = 0; k < 2; ++k) {
            const int mbase = m0 + k * 32;

            // preload all 4 B tiles so ds latency hides under the exps
            BU Bm[4];
#pragma unroll
            for (int dt = 0; dt < 4; ++dt) {
                const unsigned short* lbp =
                    buf + (dt * 16 + row) * STAGE_M + k * 32 + (hi << 3);
                Bm[dt].q[0] = *(const uint4*)(lbp);
                Bm[dt].q[1] = *(const uint4*)(lbp + 16);
            }

            // log2-domain logits -> probabilities: add, add, max, exp2
            const float4 w2a = *(const float4*)(wh2_bh + mbase + khalf);
            const float4 w2b = *(const float4*)(wh2_bh + mbase + khalf + 4);
            const float4 w2c = *(const float4*)(wh2_bh + mbase + 16 + khalf);
            const float4 w2d = *(const float4*)(wh2_bh + mbase + 16 + khalf + 4);
            const float4 v2a = *(const float4*)(wh2s_bh + mbase + khalf);
            const float4 v2b = *(const float4*)(wh2s_bh + mbase + khalf + 4);
            const float4 v2c = *(const float4*)(wh2s_bh + mbase + 16 + khalf);
            const float4 v2d = *(const float4*)(wh2s_bh + mbase + 16 + khalf + 4);

            float p[16];
            const float w2[16] = { w2a.x, w2a.y, w2a.z, w2a.w,
                                   w2b.x, w2b.y, w2b.z, w2b.w,
                                   w2c.x, w2c.y, w2c.z, w2c.w,
                                   w2d.x, w2d.y, w2d.z, w2d.w };
            const float ws2[16] = { v2a.x, v2a.y, v2a.z, v2a.w,
                                    v2b.x, v2b.y, v2b.z, v2b.w,
                                    v2c.x, v2c.y, v2c.z, v2c.w,
                                    v2d.x, v2d.y, v2d.z, v2d.w };
#pragma unroll
            for (int j = 0; j < 16; ++j)
                p[j] = __builtin_amdgcn_exp2f(fmaxf(c1 + w2[j], c2 + ws2[j]));

            union { v16bf v; unsigned u[8]; } A;
#pragma unroll
            for (int j = 0; j < 8; ++j) A.u[j] = pack2_bf16(p[2 * j], p[2 * j + 1]);

#pragma unroll
            for (int dt = 0; dt < 4; ++dt)
                acc[dt] = __builtin_amdgcn_wmma_f32_16x16x32_bf16(
                    false, A.v, false, Bm[dt].v, (short)0, acc[dt], false, false);
            // row sums: every column of accS == sum_k p(row,k)
            accS = __builtin_amdgcn_wmma_f32_16x16x32_bf16(
                false, A.v, false, Bones.v, (short)0, accS, false, false);
        }
        __syncthreads();                           // consume done before overwrite
    }

    // normalize, ELU, store. C layout: VGPR r -> row r (lanes 0-15) / r+8
    // (lanes 16-31); column = lane&15. accS[r] == S[row this lane normalizes].
    float* outb = out + ((size_t)b * GAT_N) * GAT_COLS + hh * GAT_OUT;
#pragma unroll
    for (int r = 0; r < 8; ++r) {
        const int M = r + (hi << 3);
        const float inv = 1.0f / accS[r];
#pragma unroll
        for (int dt = 0; dt < 4; ++dt) {
            const float v = acc[dt][r] * inv;
            const float o = (v > 0.f) ? v : (__expf(v) - 1.0f);
            outb[(size_t)(n0 + M) * GAT_COLS + dt * 16 + row] = o;
        }
    }
}

// ---------------------------------------------------------------------------
extern "C" void kernel_launch(void* const* d_in, const int* in_sizes, int n_in,
                              void* d_out, int out_size, void* d_ws, size_t ws_size,
                              hipStream_t stream) {
    const float* h = (const float*)d_in[0];   // (8, 2048, 128)
    const float* W = (const float*)d_in[1];   // (128, 256)
    const float* a = (const float*)d_in[2];   // (128, 4)
    float* out = (float*)d_out;               // (8, 2048, 256) f32

    char* ws = (char*)d_ws;
    // WhT bf16: 8*4*64*2048 * 2B = 8 MiB
    unsigned short* WhT = (unsigned short*)ws;
    float* Wh1  = (float*)(ws + (8u << 20));                 // 32*2048 f32
    float* Wh2  = (float*)(ws + (8u << 20) + (256u << 10));  // 32*2048 f32
    float* Wh2s = (float*)(ws + (8u << 20) + (512u << 10));  // 32*2048 f32
    float* M2   = (float*)(ws + (8u << 20) + (768u << 10));  // 32 f32

    gat_proj_kernel<<<GAT_B * GAT_N, 256, 0, stream>>>(h, W, a, WhT, Wh1, Wh2);
    gat_max_kernel<<<GAT_B * GAT_H, 256, 0, stream>>>(Wh2, M2, Wh2s);
    gat_attn_kernel<<<(GAT_B * GAT_H * (GAT_N / 16)) / 8, 256, 0, stream>>>(
        WhT, Wh1, Wh2, Wh2s, M2, out);
}
